// GAT_29411936043364
// MI455X (gfx1250) — matrix operand order repaired and checked by reading, not compiled, not run on hardware
//
#include <hip/hip_runtime.h>
#include <hip/hip_bf16.h>
#include <math.h>

typedef __bf16 bf16_t;
typedef __attribute__((ext_vector_type(16))) __bf16 v16bf;
typedef __attribute__((ext_vector_type(8)))  __bf16 v8bf;
typedef __attribute__((ext_vector_type(8)))  float  v8f;
typedef __attribute__((ext_vector_type(4)))  float  v4f;

#define H_HEADS 3
#define FDIM 64
#define LRELU_SLOPE 0.2f

__device__ __forceinline__ float elu1(float v)  { return v > 0.f ? v : __expf(v) - 1.f; }
__device__ __forceinline__ float lrelu(float v) { return v > 0.f ? v : LRELU_SLOPE * v; }

__device__ __forceinline__ void atomicMaxF(float* addr, float val) {
    int* ai  = (int*)addr;
    int  cur = __float_as_int(*addr);
    while (__int_as_float(cur) < val) {
        int prev = atomicCAS(ai, cur, __float_as_int(val));
        if (prev == cur) break;
        cur = prev;
    }
}

// ---- pack W[K,Nc] (f32, row-major) into WMMA-B fragment order (bf16) -------
// Bp[((kc*Nc + n)*32) + j] = (bf16) W[(kc*32 + j)*Nc + n]
// so each lane's B fragment (fixed col n, K-chunk kc, half hh) is two 16-byte runs
// at ((kc*Nc + n)*32 + hh*8) and 16 elements later.
__global__ void pack_b_bf16(const float* __restrict__ W, bf16_t* __restrict__ Bp,
                            int K, int Nc) {
    int i = blockIdx.x * blockDim.x + threadIdx.x;
    if (i >= K * Nc) return;
    int kc  = i >> 5;              // i / 32 within (kc,n) panel ordering below
    int j   = i & 31;
    int n   = kc % Nc;
    kc      = kc / Nc;
    Bp[i] = (bf16_t)W[(size_t)(kc * 32 + j) * Nc + n];
}

// ---- WMMA GEMM: C[M,Nc] = A[M,K](f32) * B(packed bf16) (+bias) -------------
// One wave32 computes a 16x64 strip: 4 accumulators, 4 wmma per K32 chunk.
// Requires: M%16==0, Nc%64==0, K%32==0 (all true for this model).
__global__ void wmma_gemm4(const float* __restrict__ A, const bf16_t* __restrict__ Bp,
                           float* __restrict__ C, const float* __restrict__ bias,
                           int M, int K, int Nc) {
    const int groupsN = Nc >> 6;
    const int tile    = blockIdx.x * blockDim.y + threadIdx.y;   // wave-uniform
    if (tile >= (M >> 4) * groupsN) return;
    const int tM = tile / groupsN;
    const int tg = tile % groupsN;

    const int lane = threadIdx.x;     // 0..31
    const int hh   = lane >> 4;       // lane half selects K sub-range
    const int l    = lane & 15;

    const float* __restrict__ arow = A + (size_t)(tM * 16 + l) * K;
    const int n0 = tg * 64 + l;       // column of sub-tile 0 for this lane

    v8f acc0 = {}, acc1 = {}, acc2 = {}, acc3 = {};

    const int nK = K >> 5;
#pragma unroll 2
    for (int kc = 0; kc < nK; ++kc) {
        const int kb = kc << 5;
        // A fragment: row M=l; elems 0..7 -> K=kb+8*hh.., elems 8..15 -> K=kb+16+8*hh..
        const float* p0 = arow + kb + 8 * hh;
        const float* p1 = arow + kb + 16 + 8 * hh;
        v4f a00 = *(const v4f*)(p0);
        v4f a01 = *(const v4f*)(p0 + 4);
        v4f a10 = *(const v4f*)(p1);
        v4f a11 = *(const v4f*)(p1 + 4);
        v16bf a;
#pragma unroll
        for (int i = 0; i < 4; ++i) {
            a[i]      = (bf16_t)a00[i];
            a[i + 4]  = (bf16_t)a01[i];
            a[i + 8]  = (bf16_t)a10[i];
            a[i + 12] = (bf16_t)a11[i];
        }
        // B fragments: two 16-byte runs per lane per sub-tile (K=8*hh.., K=16+8*hh..)
        const bf16_t* bbase = Bp + ((size_t)kc * Nc + n0) * 32 + hh * 8;
        v16bf b0, b1, b2, b3;
#pragma unroll
        for (int i = 0; i < 8; ++i) {
            v8bf lo, hi;
            lo = *(const v8bf*)(bbase + 0 * 512);      // sub-tile 0: cols n0
            hi = *(const v8bf*)(bbase + 0 * 512 + 16);
            b0[i] = lo[i]; b0[i + 8] = hi[i];
            lo = *(const v8bf*)(bbase + 1 * 512);      // sub-tile 1: cols n0+16
            hi = *(const v8bf*)(bbase + 1 * 512 + 16);
            b1[i] = lo[i]; b1[i + 8] = hi[i];
            lo = *(const v8bf*)(bbase + 2 * 512);      // sub-tile 2: cols n0+32
            hi = *(const v8bf*)(bbase + 2 * 512 + 16);
            b2[i] = lo[i]; b2[i + 8] = hi[i];
            lo = *(const v8bf*)(bbase + 3 * 512);      // sub-tile 3: cols n0+48
            hi = *(const v8bf*)(bbase + 3 * 512 + 16);
            b3[i] = lo[i]; b3[i + 8] = hi[i];
        }
        acc0 = __builtin_amdgcn_wmma_f32_16x16x32_bf16(false, a, false, b0, (short)0, acc0, false, false);
        acc1 = __builtin_amdgcn_wmma_f32_16x16x32_bf16(false, a, false, b1, (short)0, acc1, false, false);
        acc2 = __builtin_amdgcn_wmma_f32_16x16x32_bf16(false, a, false, b2, (short)0, acc2, false, false);
        acc3 = __builtin_amdgcn_wmma_f32_16x16x32_bf16(false, a, false, b3, (short)0, acc3, false, false);
    }

    // epilogue: M is a multiple of 16 -> no guards
    float bv0 = 0.f, bv1 = 0.f, bv2 = 0.f, bv3 = 0.f;
    if (bias) {
        bv0 = bias[n0];
        bv1 = bias[n0 + 16];
        bv2 = bias[n0 + 32];
        bv3 = bias[n0 + 48];
    }
    float* crow = C + (size_t)(tM * 16 + 8 * hh) * Nc + n0;
#pragma unroll
    for (int r = 0; r < 8; ++r) {
        float* cp = crow + (size_t)r * Nc;
        cp[0]  = acc0[r] + bv0;
        cp[16] = acc1[r] + bv1;
        cp[32] = acc2[r] + bv2;
        cp[48] = acc3[r] + bv3;
    }
}

// ---------------- el/er: per (node,head) dot of feat row with a_l/a_r -------
__global__ void eler_kernel(const float* __restrict__ feat, const float* __restrict__ al,
                            const float* __restrict__ ar, float* __restrict__ el,
                            float* __restrict__ er, int NH) {
    int i = blockIdx.x * blockDim.x + threadIdx.x;   // n*H + h
    if (i >= NH) return;
    int h = i % H_HEADS;
    const float* f   = feat + (size_t)i * FDIM;
    const float* alh = al + h * FDIM;
    const float* arh = ar + h * FDIM;
    float sl = 0.f, sr = 0.f;
#pragma unroll 8
    for (int j = 0; j < FDIM; ++j) {
        float v = f[j];
        sl += v * alh[j];
        sr += v * arh[j];
    }
    el[i] = sl;
    er[i] = sr;
}

// ---------------- init segment buffers -------------------------------------
__global__ void init_seg(float* __restrict__ maxv, float* __restrict__ sumv,
                         float* __restrict__ agg, int N) {
    int i = blockIdx.x * blockDim.x + threadIdx.x;
    int tot = N * H_HEADS * FDIM;
    if (i >= tot) return;
    agg[i] = 0.f;
    if (i < N * H_HEADS) { maxv[i] = -INFINITY; sumv[i] = 0.f; }
}

// ---------------- segment max over incoming edges ---------------------------
__global__ void edge_max(const int* __restrict__ src, const int* __restrict__ dst,
                         const float* __restrict__ el, const float* __restrict__ er,
                         float* __restrict__ maxv, int E) {
    int e = blockIdx.x * blockDim.x + threadIdx.x;
    if (e >= E) return;
    int s = src[e], d = dst[e];
#pragma unroll
    for (int h = 0; h < H_HEADS; ++h) {
        float z = lrelu(el[s * H_HEADS + h] + er[d * H_HEADS + h]);
        atomicMaxF(&maxv[d * H_HEADS + h], z);
    }
}

// ---------------- exp(e - m), store per-edge, segment-sum -------------------
__global__ void edge_expsum(const int* __restrict__ src, const int* __restrict__ dst,
                            const float* __restrict__ el, const float* __restrict__ er,
                            const float* __restrict__ maxv, float* __restrict__ sumv,
                            float* __restrict__ alpha, int E) {
    int e = blockIdx.x * blockDim.x + threadIdx.x;
    if (e >= E) return;
    int s = src[e], d = dst[e];
#pragma unroll
    for (int h = 0; h < H_HEADS; ++h) {
        float z  = lrelu(el[s * H_HEADS + h] + er[d * H_HEADS + h]);
        float ee = __expf(z - maxv[d * H_HEADS + h]);   // dst of an edge always has finite max
        alpha[(size_t)e * H_HEADS + h] = ee;
        atomicAdd(&sumv[d * H_HEADS + h], ee);
    }
}

// ---------------- normalize alpha by segment sum ----------------------------
__global__ void edge_norm(const int* __restrict__ dst, const float* __restrict__ sumv,
                          float* __restrict__ alpha, int E) {
    int e = blockIdx.x * blockDim.x + threadIdx.x;
    if (e >= E) return;
    int d = dst[e];
#pragma unroll
    for (int h = 0; h < H_HEADS; ++h) {
        float s_  = sumv[d * H_HEADS + h];
        float inv = s_ > 0.f ? 1.f / s_ : 1.f;
        alpha[(size_t)e * H_HEADS + h] *= inv;
    }
}

// ---------------- scatter-add messages: agg[dst] += feat[src] * alpha -------
__global__ void edge_agg(const int* __restrict__ src, const int* __restrict__ dst,
                         const float* __restrict__ alpha, const float* __restrict__ feat,
                         float* __restrict__ agg, int E) {
    int t = blockIdx.x * blockDim.x + threadIdx.x;   // E*64 threads
    if (t >= E * FDIM) return;
    int e = t >> 6;
    int f = t & (FDIM - 1);
    int s = src[e], d = dst[e];
#pragma unroll
    for (int h = 0; h < H_HEADS; ++h) {
        float a = alpha[(size_t)e * H_HEADS + h];
        float v = feat[((size_t)s * H_HEADS + h) * FDIM + f] * a;
        atomicAdd(&agg[((size_t)d * H_HEADS + h) * FDIM + f], v);
    }
}

// ---------------- epilogue: elu(mean_h(agg + b)) ----------------------------
__global__ void finalize_layer(const float* __restrict__ agg, const float* __restrict__ bias,
                               float* __restrict__ out, int N) {
    int i = blockIdx.x * blockDim.x + threadIdx.x;   // n*64 + f
    if (i >= N * FDIM) return;
    int n = i >> 6;
    int f = i & (FDIM - 1);
    float v = 0.f;
#pragma unroll
    for (int h = 0; h < H_HEADS; ++h)
        v += agg[((size_t)n * H_HEADS + h) * FDIM + f] + bias[h * FDIM + f];
    v *= (1.f / (float)H_HEADS);
    out[i] = elu1(v);
}

extern "C" void kernel_launch(void* const* d_in, const int* in_sizes, int n_in,
                              void* d_out, int out_size, void* d_ws, size_t ws_size,
                              hipStream_t stream) {
    const float* x   = (const float*)d_in[0];
    const int*   src = (const int*)d_in[1];
    const int*   dst = (const int*)d_in[2];
    const float* W0  = (const float*)d_in[3];
    const float* al0 = (const float*)d_in[4];
    const float* ar0 = (const float*)d_in[5];
    const float* b0  = (const float*)d_in[6];
    const float* W1  = (const float*)d_in[7];
    const float* al1 = (const float*)d_in[8];
    const float* ar1 = (const float*)d_in[9];
    const float* b1  = (const float*)d_in[10];
    const float* Wl  = (const float*)d_in[11];
    const float* bl  = (const float*)d_in[12];

    const int Din = 256;
    const int HF  = H_HEADS * FDIM;          // 192
    const int N   = in_sizes[0] / Din;       // 50000
    const int E   = in_sizes[1];             // 800000

    // ---- workspace carve-out ----
    char*  ws  = (char*)d_ws;
    size_t off = 0;
    auto carve = [&](size_t bytes) -> char* {
        char* p = ws + off;
        off += (bytes + 255) & ~(size_t)255;
        return p;
    };
    float*  feat  = (float*)carve((size_t)N * HF * 4);
    float*  agg   = (float*)carve((size_t)N * HF * 4);
    float*  alpha = (float*)carve((size_t)E * H_HEADS * 4);
    float*  el    = (float*)carve((size_t)N * H_HEADS * 4);
    float*  er    = (float*)carve((size_t)N * H_HEADS * 4);
    float*  maxv  = (float*)carve((size_t)N * H_HEADS * 4);
    float*  sumv  = (float*)carve((size_t)N * H_HEADS * 4);
    float*  hbuf  = (float*)carve((size_t)N * FDIM * 4);
    bf16_t* w0b   = (bf16_t*)carve((size_t)Din * HF * 2);
    bf16_t* w1b   = (bf16_t*)carve((size_t)FDIM * HF * 2);
    bf16_t* wlb   = (bf16_t*)carve((size_t)FDIM * FDIM * 2);

    const int T = 256;
    auto blocks = [&](long n) { return (int)((n + T - 1) / T); };

    // ---- pack weights into WMMA-B fragment order (bf16) ----
    pack_b_bf16<<<blocks(Din * HF), T, 0, stream>>>(W0, w0b, Din, HF);
    pack_b_bf16<<<blocks(FDIM * HF), T, 0, stream>>>(W1, w1b, FDIM, HF);
    pack_b_bf16<<<blocks(FDIM * FDIM), T, 0, stream>>>(Wl, wlb, FDIM, FDIM);

    auto launch_gemm = [&](const float* A, const bf16_t* B, float* C,
                           const float* bias, int M, int K, int Nc) {
        int tiles = (M / 16) * (Nc / 64);
        dim3 blk(32, 4);
        int  grid = (tiles + 3) / 4;
        wmma_gemm4<<<grid, blk, 0, stream>>>(A, B, C, bias, M, K, Nc);
    };

    auto run_edges = [&](const float* alv, const float* arv) {
        eler_kernel<<<blocks(N * H_HEADS), T, 0, stream>>>(feat, alv, arv, el, er, N * H_HEADS);
        init_seg<<<blocks((long)N * HF), T, 0, stream>>>(maxv, sumv, agg, N);
        edge_max<<<blocks(E), T, 0, stream>>>(src, dst, el, er, maxv, E);
        edge_expsum<<<blocks(E), T, 0, stream>>>(src, dst, el, er, maxv, sumv, alpha, E);
        edge_norm<<<blocks(E), T, 0, stream>>>(dst, sumv, alpha, E);
        edge_agg<<<blocks((long)E * FDIM), T, 0, stream>>>(src, dst, alpha, feat, agg, E);
    };

    // ---- layer 0: GATConv -> mean heads -> ELU ----
    launch_gemm(x, w0b, feat, nullptr, N, Din, HF);
    run_edges(al0, ar0);
    finalize_layer<<<blocks((long)N * FDIM), T, 0, stream>>>(agg, b0, hbuf, N);

    // ---- layer 1: GATConv -> mean heads -> ELU (pre-linear) ----
    launch_gemm(hbuf, w1b, feat, nullptr, N, FDIM, HF);
    run_edges(al1, ar1);
    finalize_layer<<<blocks((long)N * FDIM), T, 0, stream>>>(agg, b1, hbuf, N);

    // ---- linear head: out = elu(h) @ Wl + bl ----
    launch_gemm(hbuf, wlb, (float*)d_out, bl, N, FDIM, FDIM);
}
